// CompCSD_87978110091915
// MI455X (gfx1250) — matrix operand order repaired and checked by reading, not compiled, not run on hardware
//
#include <hip/hip_runtime.h>
#include <math.h>

#define B_   4
#define K_   256
#define HW_  16384
#define C_   64
#define L_   8
#define WAVES 8
#define TPB  (WAVES * 32)

typedef __attribute__((ext_vector_type(16))) __bf16       v16bf;
typedef __attribute__((ext_vector_type(8)))  float        v8f;
typedef __attribute__((ext_vector_type(4)))  unsigned int u32x4;
typedef __attribute__((ext_vector_type(4)))  float        f32x4;
typedef __attribute__((ext_vector_type(4)))  int          v4i;
typedef __attribute__((address_space(1)))    v4i          as1_v4i;
typedef __attribute__((address_space(3)))    v4i          as3_v4i;

// ---------------- LDS layout (bytes) ----------------
// kernelsT: 64 rows x 264 bf16 (8-elem pad -> conflict-free, 16B-aligned rows)
#define KT_ROW_B   528
#define LDS_KT     0
// per-wave A/scratch buffer: 16 rows x 264 bf16
#define S_ROW_B    528
#define S_WAVE_B   (16 * S_ROW_B)                    /* 8448 */
#define LDS_S      (64 * KT_ROW_B)                   /* 33792 */
#define LDS_CT     (LDS_S + WAVES * S_WAVE_B)        /* 101376: content cache */
#define LDS_PERM   (LDS_CT + WAVES * 16 * L_ * 4)    /* 105472 */
#define LDS_LAB    (LDS_PERM + 1024)                 /* 106496 */
#define LDS_OFF    (LDS_LAB + 1024)                  /* 107520 */
#define LDS_TOTAL  (LDS_OFF + 64)                    /* 107584 */

#if defined(__gfx1250__) && __has_builtin(__builtin_amdgcn_global_load_async_to_lds_b128)
#define HAVE_ASYNC 1
#else
#define HAVE_ASYNC 0
#endif

// ---------------- prep: label counting-sort + bf16 transposed kernels ----------------
__global__ __launch_bounds__(256)
void compcsd_prep(const float* __restrict__ kernels,
                  const int*   __restrict__ labels,
                  unsigned short* __restrict__ g_kt,   // [64][256] bf16 bits
                  int* __restrict__ g_perm,            // [256]
                  int* __restrict__ g_lab,             // [256] sorted labels
                  int* __restrict__ g_off)             // [9]
{
    __shared__ int s_perm[256];
    __shared__ int s_off[9];
    const int tid = threadIdx.x;
    if (tid == 0) {
        int cnt[L_];
        for (int l = 0; l < L_; ++l) cnt[l] = 0;
        for (int k = 0; k < K_; ++k) cnt[labels[k] & (L_ - 1)]++;
        int o = 0;
        for (int l = 0; l < L_; ++l) { s_off[l] = o; o += cnt[l]; }
        s_off[L_] = o;
        int pos[L_];
        for (int l = 0; l < L_; ++l) pos[l] = s_off[l];
        for (int k = 0; k < K_; ++k) {
            const int l = labels[k] & (L_ - 1);
            s_perm[pos[l]++] = k;                      // stable
        }
    }
    __syncthreads();
    if (tid < 256) {
        g_perm[tid] = s_perm[tid];
        g_lab[tid]  = labels[s_perm[tid]] & (L_ - 1);
    }
    if (tid < 9) g_off[tid] = s_off[tid];
    for (int i = tid; i < C_ * K_; i += 256) {
        const int c = i >> 8, kk = i & 255;
        union { __bf16 h; unsigned short u; } cv;
        cv.h = (__bf16)kernels[s_perm[kk] * C_ + c];
        g_kt[c * K_ + kk] = cv.u;                      // kernelsT[c][k'] (sorted k)
    }
}

// ---------------- fused softmax + content + scale + bf16-WMMA GEMM ----------------
__global__ __launch_bounds__(TPB)
void compcsd_main(const float* __restrict__ act,      // [B,K,H,W]
                  const unsigned short* __restrict__ g_kt,
                  const int* __restrict__ g_perm,
                  const int* __restrict__ g_lab,
                  const int* __restrict__ g_off,
                  float* __restrict__ out_content,    // [B,L,H,W]
                  float* __restrict__ out_features)   // [B,C,H,W]
{
    extern __shared__ char smem[];
    const int tid  = threadIdx.x;
    const int wave = tid >> 5;
    const int lane = tid & 31;
    const int sub  = lane & 15;   // pixel row (A) / channel col (B,D)
    const int hf   = lane >> 4;   // half selector per WMMA VGPR layouts

    char*  ktB   = smem + LDS_KT;
    char*  SB    = smem + LDS_S + wave * S_WAVE_B;
    float* Ct    = (float*)(smem + LDS_CT) + wave * (16 * L_);
    int*  s_perm = (int*)(smem + LDS_PERM);
    int*  s_lab  = (int*)(smem + LDS_LAB);
    int*  s_off  = (int*)(smem + LDS_OFF);

    // ---- stage bf16 kernelsT into padded LDS rows (async-to-LDS if available) ----
#pragma unroll
    for (int i = 0; i < 8; ++i) {
        const int row = i * 8 + wave;                 // 64 rows, 512B each
        const char* src = (const char*)g_kt + row * 512 + lane * 16;
        char*       dst = ktB + row * KT_ROW_B + lane * 16;
#if HAVE_ASYNC
        __builtin_amdgcn_global_load_async_to_lds_b128(
            (as1_v4i*)(unsigned long long)src,
            (as3_v4i*)(unsigned int)(unsigned long long)dst,
            0, 0);
#else
        *(u32x4*)dst = *(const u32x4*)src;
#endif
    }
    if (tid < 256) { s_perm[tid] = g_perm[tid]; s_lab[tid] = g_lab[tid]; }
    if (tid < 9)   { s_off[tid] = g_off[tid]; }
#if HAVE_ASYNC
#if __has_builtin(__builtin_amdgcn_s_wait_asynccnt)
    __builtin_amdgcn_s_wait_asynccnt(0);
#else
    asm volatile("s_wait_asynccnt 0" ::: "memory");
#endif
#endif
    __syncthreads();

    // ---- tile: 16 contiguous pixels of one batch image ----
    const int tile = blockIdx.x * WAVES + wave;       // 0..4095
    const int p0   = tile << 4;
    const int b    = p0 >> 14;                        // / HW_
    const int q0   = p0 & (HW_ - 1);
    const float* actb = act + (size_t)b * K_ * HW_ + q0 + sub;
    const int kb0 = hf * 128;                         // this lane's k'-range

    // Phase 1: stream column once, track max, stage bf16 (sorted-k order)
    float mx = -3.0e38f;
    for (int j = 0; j < 128; j += 2) {
        const int k0 = kb0 + j;
        const float x0 = actb[(size_t)s_perm[k0]     * HW_];
        const float x1 = actb[(size_t)s_perm[k0 + 1] * HW_];
        mx = fmaxf(mx, fmaxf(x0, x1));
        union { unsigned u; __bf16 h[2]; } pk;
        pk.h[0] = (__bf16)x0; pk.h[1] = (__bf16)x1;
        *(unsigned*)(SB + sub * S_ROW_B + k0 * 2) = pk.u;
    }
    mx = fmaxf(mx, __shfl_xor(mx, 16, 32));

    // Phase 2: exp in place (unnormalized), row sum
    float ssum = 0.f;
    for (int j = 0; j < 128; j += 2) {
        const int k0 = kb0 + j;
        unsigned* ap = (unsigned*)(SB + sub * S_ROW_B + k0 * 2);
        union { unsigned u; __bf16 h[2]; } pk; pk.u = *ap;
        const float e0 = __expf((float)pk.h[0] - mx);
        const float e1 = __expf((float)pk.h[1] - mx);
        ssum += e0 + e1;
        pk.h[0] = (__bf16)e0; pk.h[1] = (__bf16)e1;
        *ap = pk.u;
    }
    ssum += __shfl_xor(ssum, 16, 32);
    const float rinv = 1.f / ssum;

    __syncthreads();

    // Phase 3: content = contiguous segment sums (labels are sorted), f32
#pragma unroll
    for (int t = 0; t < 4; ++t) {
        const int l  = hf * 4 + t;
        const int ks = s_off[l], ke = s_off[l + 1];
        float cs = 0.f;
        for (int k = ks; k < ke; ++k)
            cs += (float)*(const __bf16*)(SB + sub * S_ROW_B + k * 2);
        const float cn = cs * rinv;
        Ct[sub * L_ + l] = cn;
        out_content[((size_t)b * L_ + l) * HW_ + q0 + sub] = cn;
    }

    __syncthreads();

    // Phase 4: scaled = vmf * content[label]  (in-place bf16 A matrix)
    for (int j = 0; j < 128; j += 2) {
        const int k0 = kb0 + j;
        unsigned* ap = (unsigned*)(SB + sub * S_ROW_B + k0 * 2);
        union { unsigned u; __bf16 h[2]; } pk; pk.u = *ap;
        const float c0 = Ct[sub * L_ + s_lab[k0]];
        const float c1 = Ct[sub * L_ + s_lab[k0 + 1]];
        pk.h[0] = (__bf16)((float)pk.h[0] * rinv * c0);
        pk.h[1] = (__bf16)((float)pk.h[1] * rinv * c1);
        *ap = pk.u;
    }

    __syncthreads();

    // Phase 5: 16x64 = A(16x256,bf16) x B(256x64,bf16), f32 accumulate
    v8f acc0 = {}, acc1 = {}, acc2 = {}, acc3 = {};
#pragma unroll
    for (int f = 0; f < 8; ++f) {
        const int kb = f * 32;
        // A 16x32 fragment: lane(sub,hf): K = kb+hf*8+[0..7] and kb+16+hf*8+[0..7]
        union { u32x4 q[2]; v16bf v; } A;
        A.q[0] = *(const u32x4*)(SB + sub * S_ROW_B + (kb + hf * 8) * 2);
        A.q[1] = *(const u32x4*)(SB + sub * S_ROW_B + (kb + 16 + hf * 8) * 2);
#define BTILE(T, ACC)                                                               \
        {                                                                           \
            union { u32x4 q[2]; v16bf v; } Bf;                                      \
            const char* kr = ktB + ((T) * 16 + sub) * KT_ROW_B + (kb + hf * 8) * 2;\
            Bf.q[0] = *(const u32x4*)(kr);                                          \
            Bf.q[1] = *(const u32x4*)(kr + 32);                                     \
            ACC = __builtin_amdgcn_wmma_f32_16x16x32_bf16(                          \
                false, A.v, false, Bf.v, (short)0, ACC, false, false);              \
        }
        BTILE(0, acc0) BTILE(1, acc1) BTILE(2, acc2) BTILE(3, acc3)
#undef BTILE
    }

    __syncthreads();

    // Phase 6: transpose D through LDS ([c][px], stride 20 f32) for coalesced stores
    float* Dt = (float*)SB;
#pragma unroll
    for (int g = 0; g < 8; ++g) {
        const int m = g + hf * 8;                     // D layout: M = vgpr + 8*half
        Dt[(0  + sub) * 20 + m] = acc0[g];
        Dt[(16 + sub) * 20 + m] = acc1[g];
        Dt[(32 + sub) * 20 + m] = acc2[g];
        Dt[(48 + sub) * 20 + m] = acc3[g];
    }
    __syncthreads();

    float* fb = out_features + (size_t)b * C_ * HW_ + q0;
#pragma unroll
    for (int i = 0; i < 8; ++i) {
        const int flat = i * 128 + lane * 4;          // 1024 f32 per wave tile
        const int c  = flat >> 4;
        const int px = flat & 15;
        const f32x4 vv = *(const f32x4*)(Dt + c * 20 + px);
        *(f32x4*)(fb + (size_t)c * HW_ + px) = vv;
    }
}

// ---------------- launch ----------------
extern "C" void kernel_launch(void* const* d_in, const int* in_sizes, int n_in,
                              void* d_out, int out_size, void* d_ws, size_t ws_size,
                              hipStream_t stream) {
    (void)in_sizes; (void)n_in; (void)out_size; (void)ws_size;
    const float* act     = (const float*)d_in[0];
    const float* kernels = (const float*)d_in[1];
    const int*   labels  = (const int*)d_in[2];

    float* out_content  = (float*)d_out;                       // B*L*HW = 524288
    float* out_features = (float*)d_out + (size_t)B_ * L_ * HW_;

    char* ws = (char*)d_ws;
    unsigned short* g_kt = (unsigned short*)ws;                // 32768 B
    int* g_perm = (int*)(ws + 32768);                          // 1024 B
    int* g_lab  = (int*)(ws + 33792);                          // 1024 B
    int* g_off  = (int*)(ws + 34816);                          // 36 B

    compcsd_prep<<<dim3(1), dim3(256), 0, stream>>>(kernels, labels, g_kt,
                                                    g_perm, g_lab, g_off);

    (void)hipFuncSetAttribute((const void*)compcsd_main,
                              hipFuncAttributeMaxDynamicSharedMemorySize,
                              LDS_TOTAL);
    const int tiles  = (B_ * HW_) / 16;                        // 4096
    const int blocks = tiles / WAVES;                          // 512
    compcsd_main<<<dim3(blocks), dim3(TPB), LDS_TOTAL, stream>>>(
        act, g_kt, g_perm, g_lab, g_off, out_content, out_features);
}
